// BAM_22325240004878
// MI455X (gfx1250) — matrix operand order repaired and checked by reading, not compiled, not run on hardware
//
#include <hip/hip_runtime.h>

// ---------------------------------------------------------------------------
// Self-attention (SAGAN style), MI455X / gfx1250, wave32, f16 WMMA + async LDS
// B=8, C=256, CK=32, N=H*W=4096
// ---------------------------------------------------------------------------

typedef _Float16 h16v __attribute__((ext_vector_type(16)));
typedef _Float16 h8v  __attribute__((ext_vector_type(8)));
typedef float    f8v  __attribute__((ext_vector_type(8)));
typedef float    f4v  __attribute__((ext_vector_type(4)));

#define NB   8
#define NC   256
#define NCK  32
#define NN   4096
#define KC   64        // keys per chunk in the attention loop
#define NCH  (NN / KC) // 64 chunks

__device__ __forceinline__ f8v wmma_f16(h16v a, h16v b, f8v c) {
  return __builtin_amdgcn_wmma_f32_16x16x32_f16(false, a, false, b, (short)0, c,
                                                false, false);
}

__device__ __forceinline__ h16v mk16(const _Float16* p) {
  h8v lo = *(const h8v*)p;
  h8v hi = *(const h8v*)(p + 8);
  return __builtin_shufflevector(lo, hi, 0, 1, 2, 3, 4, 5, 6, 7,
                                 8, 9, 10, 11, 12, 13, 14, 15);
}

// ---------------------------------------------------------------------------
// Kernel 1: q/k/v projections via f16 WMMA, f32 accumulate.
//   qT: [B][N][32] f16 (position-major, A-layout friendly)
//   kT: [B][N][32] f16
//   v : [B][C][N]  f16 (channel-major, B-layout friendly for PV GEMM)
// Waves 0..8191   : V, 4 o-tiles each  (b = w>>10, og = (w>>8)&3, nt = w&255)
// Waves 8192..10239: Q+K, 2+2 o-tiles  (b = w2>>8, nt = w2&255)
// ---------------------------------------------------------------------------
__global__ __launch_bounds__(256) void bam_proj_kernel(
    const float* __restrict__ x,
    const float* __restrict__ Wq, const float* __restrict__ bq,
    const float* __restrict__ Wk, const float* __restrict__ bk,
    const float* __restrict__ Wv, const float* __restrict__ bv,
    _Float16* __restrict__ qT, _Float16* __restrict__ kT,
    _Float16* __restrict__ vO) {
  const int wave = (int)((blockIdx.x * blockDim.x + threadIdx.x) >> 5);
  const int lane = (int)(threadIdx.x & 31);
  const int gh = lane >> 4;   // lane half (0/1)
  const int lm = lane & 15;

  const bool isV = wave < 8192;
  int b, nt, og = 0;
  if (isV) { b = wave >> 10; og = (wave >> 8) & 3; nt = wave & 255; }
  else     { int w2 = wave - 8192; b = w2 >> 8; nt = w2 & 255; }
  const int n0 = nt * 16;

  f8v acc[4];
#pragma unroll
  for (int t = 0; t < 4; ++t)
#pragma unroll
    for (int r = 0; r < 8; ++r) acc[t][r] = 0.0f;

  const float* xb = x + (size_t)b * NC * NN;

  for (int kk = 0; kk < 8; ++kk) {
    const int c0 = kk * 32;
    // B tile: b[i] = x[c0 + (i&7) + 16*(i>>3) + 8*gh][n0 + lm]  (coalesced across lanes)
    h16v bt;
#pragma unroll
    for (int i = 0; i < 16; ++i)
      bt[i] = (_Float16)xb[(size_t)(c0 + (i & 7) + 16 * (i >> 3) + 8 * gh) * NN + n0 + lm];
#pragma unroll
    for (int t = 0; t < 4; ++t) {
      const float* W;
      int o0;
      if (isV) { W = Wv; o0 = og * 64 + t * 16; }
      else     { W = (t < 2) ? Wq : Wk; o0 = (t & 1) * 16; }
      // A tile: a[i] = W[o0+lm][c0 + (i&7) + 16*(i>>3) + 8*gh]
      const float* wr = W + (size_t)(o0 + lm) * NC + c0;
      f4v a0 = *(const f4v*)(wr + 8 * gh);
      f4v a1 = *(const f4v*)(wr + 8 * gh + 4);
      f4v a2 = *(const f4v*)(wr + 16 + 8 * gh);
      f4v a3 = *(const f4v*)(wr + 20 + 8 * gh);
      h16v at;
#pragma unroll
      for (int i = 0; i < 4; ++i) {
        at[i]      = (_Float16)a0[i];
        at[i + 4]  = (_Float16)a1[i];
        at[i + 8]  = (_Float16)a2[i];
        at[i + 12] = (_Float16)a3[i];
      }
      acc[t] = wmma_f16(at, bt, acc[t]);
    }
  }

  // Bias + store. D layout: m = r + 8*gh (rel. o0), n = lm (rel. n0)
#pragma unroll
  for (int t = 0; t < 4; ++t) {
    if (isV) {
      const int o0 = og * 64 + t * 16;
      _Float16* vout = vO + (size_t)b * NC * NN;
#pragma unroll
      for (int r = 0; r < 8; ++r) {
        const int o = o0 + 8 * gh + r;
        vout[(size_t)o * NN + n0 + lm] = (_Float16)(acc[t][r] + bv[o]);
      }
    } else {
      const int o0 = (t & 1) * 16;
      const float* bias = (t < 2) ? bq : bk;
      _Float16* outp = ((t < 2) ? qT : kT) + (size_t)b * NN * NCK;
      h8v pk;
#pragma unroll
      for (int r = 0; r < 8; ++r)
        pk[r] = (_Float16)(acc[t][r] + bias[o0 + 8 * gh + r]);
      *(h8v*)(outp + (size_t)(n0 + lm) * NCK + o0 + 8 * gh) = pk;
    }
  }
}

// ---------------------------------------------------------------------------
// Async chunk prefetch: 64 keys -> LDS.
//   K: 64 rows x 64B = 4KB  -> 1 b128 per thread
//   V: 256 rows x 128B = 32KB -> 8 b128 per thread
// 9 uniform async ops per thread per chunk.
// ---------------------------------------------------------------------------
__device__ __forceinline__ void issue_chunk(int tid, uint32_t skOff,
                                            uint32_t svOff, uint64_t kBase,
                                            uint64_t vBase, int n0) {
  {
    uint32_t kg = (uint32_t)((n0 + (tid >> 2)) * (NCK * 2) + (tid & 3) * 16);
    uint32_t kl = skOff + (uint32_t)tid * 16;
    asm volatile("global_load_async_to_lds_b128 %0, %1, %2"
                 :: "v"(kl), "v"(kg), "s"(kBase) : "memory");
  }
#pragma unroll
  for (int it = 0; it < 8; ++it) {
    int idx = tid + 256 * it;                 // 2048 b128 slots
    uint32_t vg = (uint32_t)((idx >> 3) * (NN * 2) + n0 * 2 + (idx & 7) * 16);
    uint32_t vl = svOff + (uint32_t)idx * 16;
    asm volatile("global_load_async_to_lds_b128 %0, %1, %2"
                 :: "v"(vl), "v"(vg), "s"(vBase) : "memory");
  }
}

// ---------------------------------------------------------------------------
// Kernel 2: fused flash attention + gamma*out + x residual.
// Grid: 8 batches * 32 blocks (128 queries each), 256 threads (8 waves).
// Each wave: 16 query rows x 256 channels, private online softmax,
// per-lane partial row-sums (reduced once in the epilogue).
// ---------------------------------------------------------------------------
__global__ __launch_bounds__(256) void bam_attn_kernel(
    const _Float16* __restrict__ qT, const _Float16* __restrict__ kT,
    const _Float16* __restrict__ vO, const float* __restrict__ x,
    const float* __restrict__ gamma_p, float* __restrict__ out) {
  __shared__ __align__(16) _Float16 sK[2][KC * NCK];   //  2x 4KB  [key][c]
  __shared__ __align__(16) _Float16 sV[2][NC * KC];    //  2x32KB  [c][key]
  __shared__ __align__(16) _Float16 sP[8][16 * KC];    //  8x 2KB  [m][key]

  const int tid = (int)threadIdx.x;
  const int wv = tid >> 5;
  const int lane = tid & 31;
  const int gh = lane >> 4;
  const int lm = lane & 15;

  const int b = (int)(blockIdx.x >> 5);
  const int mblk = (int)(blockIdx.x & 31);
  const int mbase = mblk * 128 + wv * 16;

  // Resident Q A-tile: a[i] = qT[b][mbase+lm][(i&7)+16*(i>>3)+8*gh]
  const _Float16* qrow = qT + ((size_t)b * NN + mbase + lm) * NCK;
  const h16v qa = mk16(qrow + 8 * gh);  // careful: needs split halves
  // (mk16 reads p..p+15 contiguous; A-layout needs [8g..8g+7] and [16+8g..]);
  // rebuild properly:
  const h16v qa2 = __builtin_shufflevector(
      *(const h8v*)(qrow + 8 * gh), *(const h8v*)(qrow + 16 + 8 * gh),
      0, 1, 2, 3, 4, 5, 6, 7, 8, 9, 10, 11, 12, 13, 14, 15);
  (void)qa;

  f8v acc[16];
#pragma unroll
  for (int t = 0; t < 16; ++t)
#pragma unroll
    for (int r = 0; r < 8; ++r) acc[t][r] = 0.0f;

  float mx[8], lpart[8];
#pragma unroll
  for (int r = 0; r < 8; ++r) { mx[r] = -3.0e38f; lpart[r] = 0.0f; }

  f8v zero;
#pragma unroll
  for (int r = 0; r < 8; ++r) zero[r] = 0.0f;

  const uint64_t kBase = (uint64_t)(uintptr_t)(kT + (size_t)b * NN * NCK);
  const uint64_t vBase = (uint64_t)(uintptr_t)(vO + (size_t)b * NC * NN);
  const uint32_t skOff[2] = {(uint32_t)(uintptr_t)&sK[0][0],
                             (uint32_t)(uintptr_t)&sK[1][0]};
  const uint32_t svOff[2] = {(uint32_t)(uintptr_t)&sV[0][0],
                             (uint32_t)(uintptr_t)&sV[1][0]};

  issue_chunk(tid, skOff[0], svOff[0], kBase, vBase, 0);   // prime chunk 0

  for (int ch = 0; ch < NCH; ++ch) {
    const int cur = ch & 1;
    const int nxt = cur ^ 1;
    // Prefetch next chunk into the buffer everyone finished reading last iter.
    issue_chunk(tid, skOff[nxt], svOff[nxt], kBase, vBase,
                ((ch + 1) & (NCH - 1)) * KC);
    // Wait for previous chunk's 9 ops (leave the 9 just issued in flight).
    asm volatile("s_wait_asynccnt 0x9" ::: "memory");
    __syncthreads();   // chunk `cur` visible to all waves

    // ---- S = Q * K^T : 16 rows x 64 keys (4 WMMAs) ----
    f8v s[4];
#pragma unroll
    for (int j = 0; j < 4; ++j) {
      const _Float16* kp = &sK[cur][(size_t)(16 * j + lm) * NCK + 8 * gh];
      h16v kb = __builtin_shufflevector(
          *(const h8v*)kp, *(const h8v*)(kp + 16),
          0, 1, 2, 3, 4, 5, 6, 7, 8, 9, 10, 11, 12, 13, 14, 15);
      s[j] = wmma_f16(qa2, kb, zero);
    }

    // ---- Online softmax; write P (f16) to private LDS tile ----
    _Float16* prow = sP[wv];
    bool upd = false;
    float sc[8];
#pragma unroll
    for (int r = 0; r < 8; ++r) {
      float cm = fmaxf(fmaxf(s[0][r], s[1][r]), fmaxf(s[2][r], s[3][r]));
      cm = fmaxf(cm, __shfl_xor(cm, 1, 32));
      cm = fmaxf(cm, __shfl_xor(cm, 2, 32));
      cm = fmaxf(cm, __shfl_xor(cm, 4, 32));
      cm = fmaxf(cm, __shfl_xor(cm, 8, 32));
      const float nm = fmaxf(mx[r], cm);
      sc[r] = __expf(mx[r] - nm);
      upd = upd || (nm > mx[r]);
      mx[r] = nm;
      const float p0 = __expf(s[0][r] - nm);
      const float p1 = __expf(s[1][r] - nm);
      const float p2 = __expf(s[2][r] - nm);
      const float p3 = __expf(s[3][r] - nm);
      // Per-lane partial row sum; reduced across lanes once at the end.
      lpart[r] = lpart[r] * sc[r] + ((p0 + p1) + (p2 + p3));
      const int m = r + 8 * gh;
      prow[m * KC + lm]      = (_Float16)p0;
      prow[m * KC + 16 + lm] = (_Float16)p1;
      prow[m * KC + 32 + lm] = (_Float16)p2;
      prow[m * KC + 48 + lm] = (_Float16)p3;
    }
    // Rescale accumulators only when the running max actually moved.
    if (__any(upd)) {
#pragma unroll
      for (int t = 0; t < 16; ++t)
#pragma unroll
        for (int r = 0; r < 8; ++r) acc[t][r] *= sc[r];
    }

    // ---- P A-tiles back from LDS (transposed layout) ----
    const _Float16* pr = sP[wv] + (size_t)lm * KC;
    const h16v pa0 = __builtin_shufflevector(
        *(const h8v*)(pr + 8 * gh), *(const h8v*)(pr + 16 + 8 * gh),
        0, 1, 2, 3, 4, 5, 6, 7, 8, 9, 10, 11, 12, 13, 14, 15);
    const h16v pa1 = __builtin_shufflevector(
        *(const h8v*)(pr + 32 + 8 * gh), *(const h8v*)(pr + 48 + 8 * gh),
        0, 1, 2, 3, 4, 5, 6, 7, 8, 9, 10, 11, 12, 13, 14, 15);

    // ---- O += P * V^T : 32 WMMAs over the 256 output channels ----
#pragma unroll
    for (int t = 0; t < 16; ++t) {
      const _Float16* vp = &sV[cur][(size_t)(t * 16 + lm) * KC + 8 * gh];
      h16v vb0 = __builtin_shufflevector(
          *(const h8v*)vp, *(const h8v*)(vp + 16),
          0, 1, 2, 3, 4, 5, 6, 7, 8, 9, 10, 11, 12, 13, 14, 15);
      h16v vb1 = __builtin_shufflevector(
          *(const h8v*)(vp + 32), *(const h8v*)(vp + 48),
          0, 1, 2, 3, 4, 5, 6, 7, 8, 9, 10, 11, 12, 13, 14, 15);
      acc[t] = wmma_f16(pa0, vb0, acc[t]);
      acc[t] = wmma_f16(pa1, vb1, acc[t]);
    }
    __syncthreads();   // all waves done with buffer `cur` before it is reused
  }

  // ---- Epilogue: reduce row sums, out = gamma * (acc / l) + x ----
  const float gmv = gamma_p[0];
  float rinv[8];
#pragma unroll
  for (int r = 0; r < 8; ++r) {
    float lr = lpart[r];
    lr += __shfl_xor(lr, 1, 32);
    lr += __shfl_xor(lr, 2, 32);
    lr += __shfl_xor(lr, 4, 32);
    lr += __shfl_xor(lr, 8, 32);
    rinv[r] = 1.0f / lr;
  }

  const float* xb = x + (size_t)b * NC * NN;
  float* ob = out + (size_t)b * NC * NN;
  const int m0 = mbase + 8 * gh;
#pragma unroll
  for (int t = 0; t < 16; ++t) {
    const int c = t * 16 + lm;
    const float* xp = xb + (size_t)c * NN + m0;
    float* op = ob + (size_t)c * NN + m0;
    f4v x0 = *(const f4v*)xp;
    f4v x1 = *(const f4v*)(xp + 4);
    f4v o0, o1;
#pragma unroll
    for (int r = 0; r < 4; ++r) {
      o0[r] = gmv * acc[t][r] * rinv[r] + x0[r];
      o1[r] = gmv * acc[t][r + 4] * rinv[r + 4] + x1[r];
    }
    *(f4v*)op = o0;
    *(f4v*)(op + 4) = o1;
  }
}

// ---------------------------------------------------------------------------
extern "C" void kernel_launch(void* const* d_in, const int* in_sizes, int n_in,
                              void* d_out, int out_size, void* d_ws,
                              size_t ws_size, hipStream_t stream) {
  (void)in_sizes; (void)n_in; (void)out_size; (void)ws_size;
  const float* x  = (const float*)d_in[0];
  const float* Wq = (const float*)d_in[1];
  const float* bq = (const float*)d_in[2];
  const float* Wk = (const float*)d_in[3];
  const float* bk = (const float*)d_in[4];
  const float* Wv = (const float*)d_in[5];
  const float* bv = (const float*)d_in[6];
  const float* gm = (const float*)d_in[7];
  float* out = (float*)d_out;

  _Float16* qT = (_Float16*)d_ws;                         // 2 MB
  _Float16* kT = qT + (size_t)NB * NN * NCK;              // 2 MB
  _Float16* vO = kT + (size_t)NB * NN * NCK;              // 16 MB

  // 10240 waves: 8192 for V tiles, 2048 for Q+K tiles.
  bam_proj_kernel<<<1280, 256, 0, stream>>>(x, Wq, bq, Wk, bk, Wv, bv,
                                            qT, kT, vO);
  // 8 batches * 32 query-blocks of 128 rows.
  bam_attn_kernel<<<NB * 32, 256, 0, stream>>>(qT, kT, vO, x, gm, out);
}
